// TriangleAttention_23124103922386
// MI455X (gfx1250) — matrix operand order repaired and checked by reading, not compiled, not run on hardware
//
#include <hip/hip_runtime.h>
#include <math.h>

// ---------------- problem constants ----------------
constexpr int NRES   = 320;   // residues
constexpr int DIMD   = 128;   // pairwise channel dim
constexpr int NHEADS = 4;
constexpr int DHEAD  = 32;
constexpr int NKT    = 20;    // 320 / 16 tiles
constexpr int WPB    = 8;     // waves per block (256 threads, wave32)
#define NEGBIG (-1.0e9f)

// ---------------- WMMA types ----------------
typedef __attribute__((ext_vector_type(16))) _Float16     v16h;
typedef __attribute__((ext_vector_type(8)))  float        v8f;
typedef __attribute__((ext_vector_type(4)))  unsigned int v4u;
union V16H { v16h v; v4u q[2]; };

__device__ __forceinline__ v8f zero8() {
  v8f z = {0.f, 0.f, 0.f, 0.f, 0.f, 0.f, 0.f, 0.f};
  return z;
}

__device__ __forceinline__ v8f wmma16(v16h a, v16h b, v8f c) {
  // D = A(16x32 f16) * B(32x16 f16) + C(16x16 f32)
  return __builtin_amdgcn_wmma_f32_16x16x32_f16(false, a, false, b, (short)0, c,
                                                false, false);
}

// ---------------- LDS layout (dynamic shared); all strides multiple of 8 f16 (16B) ----
constexpr int SK_STRIDE  = 136;  // K resident: [320][128] f16 row-major (k, dh)
constexpr int SVT_STRIDE = 328;  // V resident TRANSPOSED: [128][320] f16 (dh, k)
constexpr int SW_STRIDE  = 136;  // weight slabs TRANSPOSED: [128 n][128 k] f16
constexpr int SP_STRIDE  = 328;  // softmax probs: [16][320] f16
constexpr int SQ_STRIDE  = 40;   // q tile: [16][32] f16
constexpr int SXT_STRIDE = 136;  // X tile / gated tile: [16][128] f16

constexpr int OFF_SK   = 0;
constexpr int OFF_SVT  = OFF_SK  + NRES * SK_STRIDE;       // 43520
constexpr int OFF_SW   = OFF_SVT + 128 * SVT_STRIDE;       // 85504
constexpr int OFF_SP   = OFF_SW  + 3 * 128 * SW_STRIDE;    // 137728
constexpr int OFF_SQ   = OFF_SP  + 16 * SP_STRIDE;         // 142976
constexpr int OFF_SXT  = OFF_SQ  + 16 * SQ_STRIDE;         // 143616
constexpr int HALF_TOT = OFF_SXT + 16 * SXT_STRIDE;        // 145792 halfs
constexpr int FOFF_RED = 16 * DIMD;                        // after sAcc
constexpr int FOFF_SUM = FOFF_RED + WPB * 16;
constexpr int FLT_TOT  = FOFF_SUM + NHEADS * 16;           // 2240 floats
constexpr size_t SMEM_BYTES = (size_t)HALF_TOT * 2 + (size_t)FLT_TOT * 4; // 300544

// ---------------- wave32 xor reductions via ds_swizzle ----------------
#define SWZ_XOR_F(x, m) \
  __int_as_float(__builtin_amdgcn_ds_swizzle(__float_as_int(x), (0x1f | ((m) << 10))))

__device__ __forceinline__ float redmax16(float x) {  // across 16-lane half
  x = fmaxf(x, SWZ_XOR_F(x, 1));
  x = fmaxf(x, SWZ_XOR_F(x, 2));
  x = fmaxf(x, SWZ_XOR_F(x, 4));
  x = fmaxf(x, SWZ_XOR_F(x, 8));
  return x;
}
__device__ __forceinline__ float redsum16(float x) {
  x += SWZ_XOR_F(x, 1);
  x += SWZ_XOR_F(x, 2);
  x += SWZ_XOR_F(x, 4);
  x += SWZ_XOR_F(x, 8);
  return x;
}

// ---------------- fragment loaders (ISA 7.12.2 layouts), wide LDS reads ----------------
// A 16x32 f16: lane laneN = row M; needs K = {0..7}+laneH*8 and {16..23}+laneH*8
// -> two contiguous 16B chunks at +0 and +32B  => 2x ds_load_b128
__device__ __forceinline__ v16h load_a_16x32(const _Float16* m, int stride, int k0,
                                             int laneN, int laneH) {
  V16H u;
  const v4u* p = reinterpret_cast<const v4u*>(m + laneN * stride + k0 + laneH * 8);
  u.q[0] = p[0];
  u.q[1] = p[2];
  return u.v;
}

// B 32x16 f16 where the LDS matrix is stored as the TRANSPOSE of B:
// B[kd][n] = m[(row0 + n)*stride + c0 + kd]; per lane K = {0..7}+laneH*8 and {16..23}+laneH*8
// => 2x ds_load_b128
__device__ __forceinline__ v16h load_bt_32x16(const _Float16* m, int stride, int row0,
                                              int c0, int laneN, int laneH) {
  V16H u;
  const v4u* p =
      reinterpret_cast<const v4u*>(m + (row0 + laneN) * stride + c0 + laneH * 8);
  u.q[0] = p[0];
  u.q[1] = p[2];
  return u.v;
}

// stage a 128x128 fp32 weight matrix TRANSPOSED into LDS f16: dst[n][k] = src[k][n]
__device__ __forceinline__ void stage_wt(_Float16* dst, const float* __restrict__ src,
                                         int tid) {
  for (int e = tid; e < 128 * 128; e += 256) {
    int k = e >> 7, n = e & 127;                 // global read coalesced
    dst[n * SW_STRIDE + k] = (_Float16)src[e];   // LDS write: 68-dword lane stride, conflict-free
  }
}

// ================= kernel A: bias[h][j][k] = sum_d pair[j,k,d] * Wb[d,h] =================
__global__ __launch_bounds__(256, 1)
void tri_bias_kernel(const float* __restrict__ X, const float* __restrict__ Wb,
                     float* __restrict__ biasT) {
  const int j = blockIdx.x;
  const int tid = threadIdx.x, w = tid >> 5, l = tid & 31;
  float wb[4][NHEADS];
#pragma unroll
  for (int q = 0; q < 4; ++q)
#pragma unroll
    for (int h = 0; h < NHEADS; ++h) wb[q][h] = Wb[(l + 32 * q) * NHEADS + h];

  for (int k = w; k < NRES; k += WPB) {
    const float* xp = X + ((size_t)j * NRES + k) * DIMD;
    float acc[NHEADS] = {0.f, 0.f, 0.f, 0.f};
#pragma unroll
    for (int q = 0; q < 4; ++q) {
      float x = xp[l + 32 * q];
#pragma unroll
      for (int h = 0; h < NHEADS; ++h) acc[h] += x * wb[q][h];
    }
#pragma unroll
    for (int h = 0; h < NHEADS; ++h) {
      float a = acc[h];
      a += SWZ_XOR_F(a, 1);
      a += SWZ_XOR_F(a, 2);
      a += SWZ_XOR_F(a, 4);
      a += SWZ_XOR_F(a, 8);
      a += SWZ_XOR_F(a, 16);
      acc[h] = a;
    }
    if (l == 0) {
#pragma unroll
      for (int h = 0; h < NHEADS; ++h)
        biasT[((size_t)h * NRES + j) * NRES + k] = acc[h];
    }
  }
}

// ================= kernel B: fused triangle attention, one block per row i =================
__global__ __launch_bounds__(256, 1)
void tri_attn_kernel(const float* __restrict__ X, const unsigned char* __restrict__ mask,
                     const float* __restrict__ Wq, const float* __restrict__ Wk,
                     const float* __restrict__ Wv, const float* __restrict__ Wg,
                     const float* __restrict__ Wo, const float* __restrict__ biasT,
                     float* __restrict__ out) {
  extern __shared__ char smem_raw[];
  _Float16* sH  = (_Float16*)smem_raw;
  _Float16* sK  = sH + OFF_SK;    // [320][136] f16  K = X_i Wk   (k, dh)
  _Float16* sVT = sH + OFF_SVT;   // [128][328] f16  V^T          (dh, k)
  _Float16* sW  = sH + OFF_SW;    // 3 x [128][136] f16 transposed weight slabs
  _Float16* sP  = sH + OFF_SP;    // [16][328] f16 softmax probs
  _Float16* sQ  = sH + OFF_SQ;    // [16][40]  f16 q tile (pre-scaled)
  _Float16* sXt = sH + OFF_SXT;   // [16][136] f16 X tile / gated tile
  float* sF   = (float*)(sH + HALF_TOT);
  float* sAcc = sF;               // [16][128] f32 attention output accumulator
  float* sRed = sF + FOFF_RED;    // [8][16]   f32 cross-wave reduction scratch
  float* sSum = sF + FOFF_SUM;    // [4][16]   f32 softmax denominators per head

  const int i = blockIdx.x;
  const int tid = threadIdx.x;
  const int w = tid >> 5;
  const int lane = tid & 31;
  const int laneN = lane & 15;
  const int laneH = lane >> 4;
  const float scale = 0.17677669529663687f;  // 1/sqrt(32)

  // ---- stage Wk^T, Wv^T (slots 0,1) ----
  stage_wt(sW + 0 * 128 * SW_STRIDE, Wk, tid);
  stage_wt(sW + 1 * 128 * SW_STRIDE, Wv, tid);
  __syncthreads();

  // ---- phase 1: K (row-major) and V^T projections, resident f16 in LDS ----
  for (int kt = 0; kt < NKT; ++kt) {
    // load X tile with b128 global loads
    for (int e = tid; e < (16 * DIMD) / 4; e += 256) {
      int base = e * 4;
      int r = base >> 7, d = base & 127;
      const float4 x =
          *reinterpret_cast<const float4*>(X + ((size_t)i * NRES + kt * 16 + r) * DIMD + d);
      _Float16* dst = sXt + r * SXT_STRIDE + d;
      dst[0] = (_Float16)x.x;
      dst[1] = (_Float16)x.y;
      dst[2] = (_Float16)x.z;
      dst[3] = (_Float16)x.w;
    }
    __syncthreads();
    v8f ck = zero8(), cv = zero8();
    const int n0 = w * 16;  // wave owns 16 of the 128 inner columns
#pragma unroll
    for (int kb = 0; kb < 4; ++kb) {
      v16h a  = load_a_16x32(sXt, SXT_STRIDE, kb * 32, laneN, laneH);
      v16h bk = load_bt_32x16(sW + 0 * 128 * SW_STRIDE, SW_STRIDE, n0, kb * 32, laneN, laneH);
      ck = wmma16(a, bk, ck);
      v16h bv = load_bt_32x16(sW + 1 * 128 * SW_STRIDE, SW_STRIDE, n0, kb * 32, laneN, laneH);
      cv = wmma16(a, bv, cv);
    }
#pragma unroll
    for (int r = 0; r < 8; ++r) {
      int row = kt * 16 + r + laneH * 8;  // k index
      sK[row * SK_STRIDE + n0 + laneN] = (_Float16)ck[r];
      sVT[(n0 + laneN) * SVT_STRIDE + row] = (_Float16)cv[r];  // transposed store
    }
    __syncthreads();
  }

  // ---- restage Wq^T, Wg^T, Wo^T (slots 0,1,2) ----
  stage_wt(sW + 0 * 128 * SW_STRIDE, Wq, tid);
  stage_wt(sW + 1 * 128 * SW_STRIDE, Wg, tid);
  stage_wt(sW + 2 * 128 * SW_STRIDE, Wo, tid);
  __syncthreads();

  // ---- phase 2: per j-tile attention + gate + output projection ----
  for (int jt = 0; jt < NKT; ++jt) {
    const int j0 = jt * 16;
    for (int e = tid; e < (16 * DIMD) / 4; e += 256) {
      int base = e * 4;
      int r = base >> 7, d = base & 127;
      const float4 x =
          *reinterpret_cast<const float4*>(X + ((size_t)i * NRES + j0 + r) * DIMD + d);
      _Float16* dst = sXt + r * SXT_STRIDE + d;
      dst[0] = (_Float16)x.x;
      dst[1] = (_Float16)x.y;
      dst[2] = (_Float16)x.z;
      dst[3] = (_Float16)x.w;
      float* az = sAcc + base;
      az[0] = 0.f; az[1] = 0.f; az[2] = 0.f; az[3] = 0.f;
    }
    __syncthreads();

    for (int h = 0; h < NHEADS; ++h) {
      // --- q tile (pre-scaled): waves 0,1 each do one 16-col block ---
      if (w < 2) {
        v8f cq = zero8();
#pragma unroll
        for (int kb = 0; kb < 4; ++kb) {
          v16h a = load_a_16x32(sXt, SXT_STRIDE, kb * 32, laneN, laneH);
          v16h b = load_bt_32x16(sW, SW_STRIDE, h * DHEAD + w * 16, kb * 32, laneN, laneH);
          cq = wmma16(a, b, cq);
        }
#pragma unroll
        for (int r = 0; r < 8; ++r)
          sQ[(r + laneH * 8) * SQ_STRIDE + w * 16 + laneN] = (_Float16)(cq[r] * scale);
      }
      __syncthreads();

      v16h aQ = load_a_16x32(sQ, SQ_STRIDE, 0, laneN, laneH);

      // --- scores: wave w owns k-tiles t = w, w+8, w+16 ---
      v8f sc[3];
      int ts[3];
      int nt = 0;
      float rmaxr[8];
#pragma unroll
      for (int r = 0; r < 8; ++r) rmaxr[r] = -3.0e38f;
      for (int t = w; t < NKT; t += WPB) {
        // B = K^T: K stored row-major (k, dh) -> contiguous per lane
        v16h bK = load_bt_32x16(sK, SK_STRIDE, t * 16, h * DHEAD, laneN, laneH);
        v8f c = wmma16(aQ, bK, zero8());
        const int kcol = t * 16 + laneN;
        const bool km = (mask[kcol] != 0);
#pragma unroll
        for (int r = 0; r < 8; ++r) {
          float s = c[r] +
                    biasT[((size_t)h * NRES + (j0 + r + laneH * 8)) * NRES + kcol];
          s = km ? s : NEGBIG;
          c[r] = s;
          rmaxr[r] = fmaxf(rmaxr[r], s);
        }
        sc[nt] = c;
        ts[nt] = t;
        ++nt;
      }
      // row max: reduce within 16-lane half, then across 8 waves via LDS
#pragma unroll
      for (int r = 0; r < 8; ++r) rmaxr[r] = redmax16(rmaxr[r]);
      if (laneN == 0) {
#pragma unroll
        for (int r = 0; r < 8; ++r) sRed[w * 16 + r + laneH * 8] = rmaxr[r];
      }
      __syncthreads();
      float rowmax[8];
#pragma unroll
      for (int r = 0; r < 8; ++r) {
        float mx = -3.0e38f;
#pragma unroll
        for (int wv = 0; wv < WPB; ++wv) mx = fmaxf(mx, sRed[wv * 16 + r + laneH * 8]);
        rowmax[r] = mx;
      }
      __syncthreads();  // before reusing sRed for sums

      // exp, partial sums, write P (f16) to LDS
      float rsum[8];
#pragma unroll
      for (int r = 0; r < 8; ++r) rsum[r] = 0.f;
#pragma unroll
      for (int it = 0; it < 3; ++it) {
        if (it < nt) {
          const int t = ts[it];
          v8f c = sc[it];
#pragma unroll
          for (int r = 0; r < 8; ++r) {
            float p = __expf(c[r] - rowmax[r]);
            rsum[r] += p;
            sP[(r + laneH * 8) * SP_STRIDE + t * 16 + laneN] = (_Float16)p;
          }
        }
      }
#pragma unroll
      for (int r = 0; r < 8; ++r) rsum[r] = redsum16(rsum[r]);
      if (laneN == 0) {
#pragma unroll
        for (int r = 0; r < 8; ++r) sRed[w * 16 + r + laneH * 8] = rsum[r];
      }
      __syncthreads();
      if (w == 0 && laneN == 0) {
#pragma unroll
        for (int r = 0; r < 8; ++r) {
          float s = 0.f;
#pragma unroll
          for (int wv = 0; wv < WPB; ++wv) s += sRed[wv * 16 + r + laneH * 8];
          sSum[h * 16 + r + laneH * 8] = s;
        }
      }

      // --- PV: wave owns nb = w&1 output half; K-blocks kk = (w>>1) + 4s ---
      const int nb = w & 1;
      v8f co = zero8();
      for (int kk = (w >> 1); kk < 10; kk += 4) {
        v16h a = load_a_16x32(sP, SP_STRIDE, kk * 32, laneN, laneH);
        // B = V: stored transposed (dh, k) -> contiguous per lane
        v16h b = load_bt_32x16(sVT, SVT_STRIDE, h * DHEAD + nb * 16, kk * 32, laneN, laneH);
        co = wmma16(a, b, co);
      }
#pragma unroll
      for (int r = 0; r < 8; ++r)
        atomicAdd(&sAcc[(r + laneH * 8) * DIMD + h * DHEAD + nb * 16 + laneN], co[r]);
      __syncthreads();
    }  // heads

    // --- gate: g = sigmoid(Xt Wg); normalize, multiply, stash f16 into sXt ---
    const int n0 = w * 16;
    v8f cg = zero8();
#pragma unroll
    for (int kb = 0; kb < 4; ++kb) {
      v16h a = load_a_16x32(sXt, SXT_STRIDE, kb * 32, laneN, laneH);
      v16h b = load_bt_32x16(sW + 1 * 128 * SW_STRIDE, SW_STRIDE, n0, kb * 32, laneN, laneH);
      cg = wmma16(a, b, cg);
    }
    __syncthreads();  // everyone done reading sXt; sAcc/sSum final
    const int h2 = w >> 1;  // head of this wave's column block
#pragma unroll
    for (int r = 0; r < 8; ++r) {
      int row = r + laneH * 8;
      int col = n0 + laneN;
      float o = sAcc[row * DIMD + col] / sSum[h2 * 16 + row];
      float gate = 1.0f / (1.0f + __expf(-cg[r]));
      sXt[row * SXT_STRIDE + col] = (_Float16)(o * gate);
    }
    __syncthreads();

    // --- output projection: y = gated @ Wo, store fp32 ---
    v8f cy = zero8();
#pragma unroll
    for (int kb = 0; kb < 4; ++kb) {
      v16h a = load_a_16x32(sXt, SXT_STRIDE, kb * 32, laneN, laneH);
      v16h b = load_bt_32x16(sW + 2 * 128 * SW_STRIDE, SW_STRIDE, n0, kb * 32, laneN, laneH);
      cy = wmma16(a, b, cy);
    }
#pragma unroll
    for (int r = 0; r < 8; ++r)
      out[((size_t)i * NRES + j0 + r + laneH * 8) * DIMD + n0 + laneN] = cy[r];
    __syncthreads();
  }
}

// ================= launch =================
extern "C" void kernel_launch(void* const* d_in, const int* in_sizes, int n_in,
                              void* d_out, int out_size, void* d_ws, size_t ws_size,
                              hipStream_t stream) {
  (void)in_sizes; (void)n_in; (void)out_size; (void)ws_size;
  const float* X          = (const float*)d_in[0];
  const unsigned char* mk = (const unsigned char*)d_in[1];
  const float* Wq         = (const float*)d_in[2];
  const float* Wk         = (const float*)d_in[3];
  const float* Wv         = (const float*)d_in[4];
  const float* Wg         = (const float*)d_in[5];
  const float* Wo         = (const float*)d_in[6];
  const float* Wb         = (const float*)d_in[7];
  float* out   = (float*)d_out;
  float* biasT = (float*)d_ws;  // NHEADS*NRES*NRES floats = 1.6 MB

  hipFuncSetAttribute((const void*)tri_attn_kernel,
                      hipFuncAttributeMaxDynamicSharedMemorySize, (int)SMEM_BYTES);

  tri_bias_kernel<<<NRES, 256, 0, stream>>>(X, Wb, biasT);
  tri_attn_kernel<<<NRES, 256, SMEM_BYTES, stream>>>(X, mk, Wq, Wk, Wv, Wg, Wo,
                                                     biasT, out);
}
